// MultiTokenConceptLayer_73495480369265
// MI455X (gfx1250) — compile-verified, matrix operand order checked by reading, not run on hardware
//
#include <hip/hip_runtime.h>
#include <math.h>

// ---------------------------------------------------------------------------
// Types for CDNA5 WMMA
// ---------------------------------------------------------------------------
typedef __attribute__((ext_vector_type(8)))  __bf16 bf16x8;
typedef __attribute__((ext_vector_type(16))) __bf16 bf16x16;
typedef __attribute__((ext_vector_type(8)))  float  f32x8;

#define EPSF 1e-8f
#define D_DIM 4096
#define S_DIM 2048
#define B_DIM 4
#define DK_DIM 768
#define K_KEYS 1024
#define Q_DIM 256
#define MAX_TOK 4
#define OUT_S (S_DIM + MAX_TOK)   // 2052

#if __has_builtin(__builtin_amdgcn_s_wait_asynccnt)
#define ASYNC_WAIT(n) __builtin_amdgcn_s_wait_asynccnt(n)
#else
#define ASYNC_WAIT(n) asm volatile("s_wait_asynccnt %0" ::"n"(n) : "memory")
#endif

// ---------------------------------------------------------------------------
// Kernel 1: fp32 GEMM  out[b,s,e] = sum_d hidden[b,s,d] * W[e,d] + bias[e]
// Split-bf16 (hi+lo) WMMA: hi*hi + hi*lo + lo*hi, fp32 accumulate.
// Double-buffered fp32 tiles in LDS filled by GLOBAL_LOAD_ASYNC_TO_LDS_B128
// (ASYNCcnt), XOR-swizzled 16B chunks for bank-conflict-free ds_load_b128.
// WG = 256 threads (8 waves), tile 128(M) x 128(N), K-step 32.
// ---------------------------------------------------------------------------

// Issue 4 async 16B global->LDS DMAs covering one 128x32 fp32 tile slice.
// Logical chunk c (of 8 per row) is stored at physical chunk c ^ ((row>>1)&7).
__device__ __forceinline__ void stage_async(const float* __restrict__ gbase, int grow0,
                                            unsigned lds_base_off, int k0, int tid) {
#pragma unroll
    for (int i = 0; i < 4; ++i) {
        int idx = tid + (i << 8);          // 0..1023 : 128 rows x 8 chunks
        int row = idx >> 3;
        int c   = idx & 7;
        int cs  = c ^ ((row >> 1) & 7);
        unsigned loff = lds_base_off + (unsigned)(row * 128 + cs * 16);
        unsigned goff = (unsigned)(((unsigned)(grow0 + row) * (unsigned)D_DIM +
                                    (unsigned)(k0 + c * 4)) * 4u);
        asm volatile("global_load_async_to_lds_b128 %0, %1, %2"
                     :
                     : "v"(loff), "v"(goff), "s"(gbase)
                     : "memory");
    }
}

// Load one 16x32 fragment from a swizzled fp32 LDS tile and split into
// bf16 hi/lo in the CDNA5 WMMA VGPR layout:
// lane L(0..15): row=row0+L, K={0..7,16..23}; lanes 16..31: K={8..15,24..31}
__device__ __forceinline__ void load_frag_f32(const float* base, int row0, int lane,
                                              bf16x16& hi, bf16x16& lo) {
    const int r  = row0 + (lane & 15);
    const int kb = (lane >> 4) << 3;   // 0 or 8
    const int j  = (r >> 1) & 7;
    const int c0 = kb >> 2;
    const float* rowp = base + r * 32;
    float4 f0 = *(const float4*)(rowp + (((c0 + 0) ^ j) << 2));  // K kb+0..3
    float4 f1 = *(const float4*)(rowp + (((c0 + 1) ^ j) << 2));  // K kb+4..7
    float4 f2 = *(const float4*)(rowp + (((c0 + 4) ^ j) << 2));  // K kb+16..19
    float4 f3 = *(const float4*)(rowp + (((c0 + 5) ^ j) << 2));  // K kb+20..23
    float f[16] = {f0.x, f0.y, f0.z, f0.w, f1.x, f1.y, f1.z, f1.w,
                   f2.x, f2.y, f2.z, f2.w, f3.x, f3.y, f3.z, f3.w};
#pragma unroll
    for (int t = 0; t < 16; ++t) {
        __bf16 h = (__bf16)f[t];
        hi[t] = h;
        lo[t] = (__bf16)(f[t] - (float)h);
    }
}

__global__ void __launch_bounds__(256)
gemm_split_bf16_kernel(const float* __restrict__ hs, const float* __restrict__ W,
                       const float* __restrict__ bias, float* __restrict__ out) {
    // 2 x (128x32) fp32 tiles for A and B : exactly 64 KB
    __shared__ __attribute__((aligned(16))) float ldsA[2][128 * 32];
    __shared__ __attribute__((aligned(16))) float ldsB[2][128 * 32];

    const int tid  = threadIdx.x;
    const int lane = tid & 31;
    const int wave = tid >> 5;
    const int wm   = wave >> 1;          // 0..3  -> M sub-tile of 32
    const int wn   = wave & 1;           // 0..1  -> N sub-tile of 64
    const int m0   = blockIdx.y * 128;   // row in flattened [B*S]
    const int n0   = blockIdx.x * 128;   // output feature

    unsigned a_lds[2], b_lds[2];
    a_lds[0] = (unsigned)(size_t)&ldsA[0][0];
    a_lds[1] = (unsigned)(size_t)&ldsA[1][0];
    b_lds[0] = (unsigned)(size_t)&ldsB[0][0];
    b_lds[1] = (unsigned)(size_t)&ldsB[1][0];

    // prologue: DMA first K-tile into buffer 0 (8 async ops)
    stage_async(hs, m0, a_lds[0], 0, tid);
    stage_async(W,  n0, b_lds[0], 0, tid);

    f32x8 acc[2][4] = {};

    for (int k0 = 0; k0 < D_DIM; k0 += 32) {
        const int buf = (k0 >> 5) & 1;
        if (k0 + 32 < D_DIM) {
            // DMA next tile into the other buffer while this one computes
            stage_async(hs, m0, a_lds[buf ^ 1], k0 + 32, tid);
            stage_async(W,  n0, b_lds[buf ^ 1], k0 + 32, tid);
            ASYNC_WAIT(8);   // newest 8 DMAs may stay in flight
        } else {
            ASYNC_WAIT(0);
        }
        __syncthreads();     // all waves' DMAs for `buf` landed

        const float* Abuf = &ldsA[buf][0];
        const float* Bbuf = &ldsB[buf][0];

        bf16x16 ah[2], al[2];
#pragma unroll
        for (int mi = 0; mi < 2; ++mi)
            load_frag_f32(Abuf, wm * 32 + mi * 16, lane, ah[mi], al[mi]);

#pragma unroll
        for (int ni = 0; ni < 4; ++ni) {
            bf16x16 bh, bl;
            load_frag_f32(Bbuf, wn * 64 + ni * 16, lane, bh, bl);
#pragma unroll
            for (int mi = 0; mi < 2; ++mi) {
                acc[mi][ni] = __builtin_amdgcn_wmma_f32_16x16x32_bf16(
                    false, ah[mi], false, bl, (short)0, acc[mi][ni], false, false);
                acc[mi][ni] = __builtin_amdgcn_wmma_f32_16x16x32_bf16(
                    false, al[mi], false, bh, (short)0, acc[mi][ni], false, false);
                acc[mi][ni] = __builtin_amdgcn_wmma_f32_16x16x32_bf16(
                    false, ah[mi], false, bh, (short)0, acc[mi][ni], false, false);
            }
        }
        __syncthreads();     // reads done before next iteration's DMA overwrites
    }

    // ---- epilogue: bias add + scatter to [B, 2052, 4096]
#pragma unroll
    for (int ni = 0; ni < 4; ++ni) {
        int n = n0 + wn * 64 + ni * 16 + (lane & 15);
        float bv = bias[n];
#pragma unroll
        for (int mi = 0; mi < 2; ++mi) {
            int mb = m0 + wm * 32 + mi * 16 + ((lane >> 4) << 3);
#pragma unroll
            for (int r = 0; r < 8; ++r) {
                int m  = mb + r;
                int sb = m >> 11;       // sample (m / 2048)
                int ss = m & 2047;      // row within sample
                out[(size_t)sb * (OUT_S * D_DIM) + (size_t)ss * D_DIM + n] =
                    acc[mi][ni][r] + bv;
            }
        }
    }
}

// ---------------------------------------------------------------------------
// Kernel 2: inverse L2 norms (+EPS) for keys [1024] and queries [B*Q=1024]
// ---------------------------------------------------------------------------
__global__ void norm_kernel(const float* __restrict__ keys, const float* __restrict__ cs,
                            float* __restrict__ kinv, float* __restrict__ qinv) {
    int lane = threadIdx.x & 31, wv = threadIdx.x >> 5;
    int r = blockIdx.x * 8 + wv;   // 0..2047
    const float* p = (r < K_KEYS) ? (keys + (size_t)r * DK_DIM)
                                  : (cs + (size_t)(r - K_KEYS) * DK_DIM);
    float s = 0.f;
#pragma unroll
    for (int i = 0; i < DK_DIM / 32; ++i) {
        float x = p[lane + 32 * i];
        s += x * x;
    }
#pragma unroll
    for (int off = 16; off; off >>= 1) s += __shfl_xor(s, off, 32);
    if (lane == 0) {
        float inv = 1.0f / (sqrtf(s) + EPSF);
        if (r < K_KEYS) kinv[r] = inv; else qinv[r - K_KEYS] = inv;
    }
}

// ---------------------------------------------------------------------------
// Kernel 3: per (b, 16-query slice) find lexicographic-best (sim, q, k)
// ---------------------------------------------------------------------------
__global__ void __launch_bounds__(256)
sim_kernel(const float* __restrict__ cs, const float* __restrict__ keys,
           const float* __restrict__ kinv, const float* __restrict__ qinv,
           float* __restrict__ pv, int* __restrict__ pq, int* __restrict__ pk) {
    __shared__ __attribute__((aligned(16))) float qlds[16][DK_DIM];
    __shared__ float rv[256];
    __shared__ int   rq[256];
    __shared__ int   rk[256];

    const int tid = threadIdx.x;
    const int b   = blockIdx.y;
    const int q0  = blockIdx.x * 16;

    for (int idx = tid; idx < 16 * (DK_DIM / 4); idx += 256) {
        int q = idx / (DK_DIM / 4), c4 = idx % (DK_DIM / 4);
        int gq = b * Q_DIM + q0 + q;
        float4 v = ((const float4*)(cs + (size_t)gq * DK_DIM))[c4];
        float qi = qinv[gq];
        float4 sv = {v.x * qi, v.y * qi, v.z * qi, v.w * qi};
        *(float4*)&qlds[q][c4 * 4] = sv;
    }
    __syncthreads();

    float bestv = -3.0e38f; int bq = 0, bk = 0;
    for (int k = tid; k < K_KEYS; k += 256) {
        float acc[16] = {};
        const float4* kr = (const float4*)(keys + (size_t)k * DK_DIM);
#pragma unroll 1
        for (int c4 = 0; c4 < DK_DIM / 4; ++c4) {
            float4 kv = kr[c4];
#pragma unroll
            for (int q = 0; q < 16; ++q) {
                const float* qp = &qlds[q][c4 * 4];
                acc[q] += kv.x * qp[0] + kv.y * qp[1] + kv.z * qp[2] + kv.w * qp[3];
            }
        }
        float ki = kinv[k];
#pragma unroll
        for (int q = 0; q < 16; ++q) {
            float v = acc[q] * ki;
            int gq = q0 + q;
            if (v > bestv || (v == bestv && (gq < bq || (gq == bq && k < bk)))) {
                bestv = v; bq = gq; bk = k;
            }
        }
    }

    rv[tid] = bestv; rq[tid] = bq; rk[tid] = bk;
    __syncthreads();
    for (int s = 128; s > 0; s >>= 1) {
        if (tid < s) {
            float v2 = rv[tid + s]; int q2 = rq[tid + s], k2 = rk[tid + s];
            if (v2 > rv[tid] ||
                (v2 == rv[tid] && (q2 < rq[tid] || (q2 == rq[tid] && k2 < rk[tid])))) {
                rv[tid] = v2; rq[tid] = q2; rk[tid] = k2;
            }
        }
        __syncthreads();
    }
    if (tid == 0) {
        int p = b * 16 + blockIdx.x;
        pv[p] = rv[0]; pq[p] = rq[0]; pk[p] = rk[0];
    }
}

// ---------------------------------------------------------------------------
// Kernel 4: per-sample reduce of 16 slice partials -> concept, k_tok
// ---------------------------------------------------------------------------
__global__ void pick_kernel(const float* __restrict__ pv, const int* __restrict__ pq,
                            const int* __restrict__ pk, int* __restrict__ conc,
                            int* __restrict__ ktok) {
    int b = threadIdx.x;
    if (b < B_DIM) {
        float bv = -3.0e38f; int bq = 0, bk = 0;
        for (int i = 0; i < 16; ++i) {
            int p = b * 16 + i;
            float v = pv[p]; int q = pq[p], k = pk[p];
            if (v > bv || (v == bv && (q < bq || (q == bq && k < bk)))) {
                bv = v; bq = q; bk = k;
            }
        }
        conc[b] = bk;
        float g = fminf(fmaxf(bv, 0.0f), 1.0f);
        int kt = (int)ceilf(g * (float)MAX_TOK);
        if (kt < 1) kt = 1;
        ktok[b] = kt;
    }
}

// ---------------------------------------------------------------------------
// Kernel 5: normalized, masked concept tokens -> out rows 2048..2051
// ---------------------------------------------------------------------------
__global__ void tokens_kernel(const float* __restrict__ vals, const int* __restrict__ conc,
                              const int* __restrict__ ktok, float* __restrict__ out) {
    const int tid = threadIdx.x;
    const int b = blockIdx.x >> 2, t = blockIdx.x & 3;
    const int c = conc[b], kt = ktok[b];
    const float4* src = (const float4*)(vals + ((size_t)c * MAX_TOK + t) * D_DIM);
    float4 r[4];
    float s = 0.f;
#pragma unroll
    for (int j = 0; j < 4; ++j) {
        r[j] = src[tid + j * 256];
        s += r[j].x * r[j].x + r[j].y * r[j].y + r[j].z * r[j].z + r[j].w * r[j].w;
    }
#pragma unroll
    for (int off = 16; off; off >>= 1) s += __shfl_xor(s, off, 32);
    __shared__ float wred[8];
    int lane = tid & 31, wv = tid >> 5;
    if (lane == 0) wred[wv] = s;
    __syncthreads();
    if (tid == 0) {
        float tot = 0.f;
        for (int i = 0; i < 8; ++i) tot += wred[i];
        wred[0] = 1.0f / (sqrtf(tot) + EPSF);
    }
    __syncthreads();
    float scale = (t < kt) ? wred[0] : 0.0f;
    float4* dst = (float4*)(out + (size_t)b * (OUT_S * D_DIM) + (size_t)(S_DIM + t) * D_DIM);
#pragma unroll
    for (int j = 0; j < 4; ++j) {
        float4 v = r[j];
        v.x *= scale; v.y *= scale; v.z *= scale; v.w *= scale;
        dst[tid + j * 256] = v;
    }
}

// ---------------------------------------------------------------------------
// Launch
// ---------------------------------------------------------------------------
extern "C" void kernel_launch(void* const* d_in, const int* in_sizes, int n_in,
                              void* d_out, int out_size, void* d_ws, size_t ws_size,
                              hipStream_t stream) {
    (void)in_sizes; (void)n_in; (void)out_size; (void)ws_size;
    const float* hs   = (const float*)d_in[0];  // [4,2048,4096]
    const float* cs   = (const float*)d_in[1];  // [4,256,768]
    const float* keys = (const float*)d_in[2];  // [1024,768]
    const float* vals = (const float*)d_in[3];  // [1024,4,4096]
    const float* W    = (const float*)d_in[4];  // [4096,4096]
    const float* bias = (const float*)d_in[5];  // [4096]
    float* out = (float*)d_out;                 // [4,2052,4096]

    float* wsf  = (float*)d_ws;
    float* kinv = wsf;                        // 1024 floats
    float* qinv = wsf + 1024;                 // 1024 floats
    float* pv   = wsf + 2048;                 // 64 floats
    int*   pq   = (int*)(wsf + 2048 + 64);    // 64 ints
    int*   pk   = (int*)(wsf + 2048 + 128);   // 64 ints
    int*   conc = (int*)(wsf + 2048 + 192);   // 4 ints
    int*   ktok = (int*)(wsf + 2048 + 196);   // 4 ints

    norm_kernel<<<256, 256, 0, stream>>>(keys, cs, kinv, qinv);
    sim_kernel<<<dim3(16, 4), 256, 0, stream>>>(cs, keys, kinv, qinv, pv, pq, pk);
    pick_kernel<<<1, 32, 0, stream>>>(pv, pq, pk, conc, ktok);
    tokens_kernel<<<16, 256, 0, stream>>>(vals, conc, ktok, out);
    gemm_split_bf16_kernel<<<dim3(D_DIM / 128, (B_DIM * S_DIM) / 128), 256, 0, stream>>>(
        hs, W, bias, out);
}
